// MultiHeadAttention_910533067646
// MI455X (gfx1250) — compile-verified, matrix operand order checked by reading, not run on hardware
//
#include <hip/hip_runtime.h>

// ---------------------------------------------------------------------------
// MI455X (gfx1250) fused MHA (buggy-reference-faithful) pipeline:
//   GEMM1: P  = x @ Wv^T + bv            (f32 in -> bf16 out, WMMA bf16)
//   ATTN : O  = softmax(P P^T / 8) P      per (b,h), flash-style, WMMA bf16
//   GEMM2: Y  = O @ Wo^T + bo            (bf16 in -> f32 out, WMMA bf16)
// bf16 tiles are staged into LDS with the Tensor Data Mover (TDM); TDM row
// padding reproduces the conflict-avoiding LDS strides (72 / 40 elements).
// ---------------------------------------------------------------------------

typedef __attribute__((ext_vector_type(16))) __bf16 v16bf;
typedef __attribute__((ext_vector_type(8)))  __bf16 v8bf;
typedef __attribute__((ext_vector_type(8)))  float  v8f;
typedef __attribute__((ext_vector_type(4)))  unsigned int v4u;
typedef __attribute__((ext_vector_type(8)))  int v8i;
typedef __attribute__((ext_vector_type(4)))  int v4i;

static constexpr int Bc  = 4;
static constexpr int Sc  = 2048;
static constexpr int Dc  = 1024;
static constexpr int Hc  = 16;
static constexpr int DKc = 64;
static constexpr int BSc = Bc * Sc;     // 8192 rows

// ---------------------------------------------------------------------------
// TDM: DMA a 2D bf16 tile (tile_w x tile_h, global row stride in elements)
// into LDS at lds_off, with pad_amount DWORD padding every pad-interval so the
// LDS row stride becomes tile_w + pad elements. Descriptor per ISA 8.3/8.4.
//   pad codes: interval v -> 2^(v+1) DWORDs, amount a -> (a+1) DWORDs.
// ---------------------------------------------------------------------------
__device__ __forceinline__ void tdm_load_2d_bf16(unsigned lds_off,
                                                 const void* gptr,
                                                 int tile_w, int tile_h,
                                                 int row_stride_elems,
                                                 int pad_interval_code,
                                                 int pad_amount_code)
{
  unsigned long long ga = (unsigned long long)(uintptr_t)gptr;
  v4u g0;
  g0[0] = 1u;                                            // count=1 (valid user D#)
  g0[1] = lds_off;                                       // lds_addr
  g0[2] = (unsigned)(ga & 0xFFFFFFFFu);                  // global_addr[31:0]
  g0[3] = (unsigned)((ga >> 32) & 0x1FFFFFFu) | (2u << 30); // ga[56:32] | type=2
  unsigned flags = (1u << 16)                            // data_size = 1 (2 bytes)
                 | (1u << 20)                            // pad_enable
                 | ((unsigned)pad_interval_code << 22)
                 | ((unsigned)pad_amount_code   << 25);
  v8i g1;
  g1[0] = (int)flags;                                    // wg_mask=0 | flags
  g1[1] = (int)(((unsigned)tile_w & 0xFFFFu) << 16);     // tensor_dim0[15:0]
  g1[2] = (int)((((unsigned)tile_w >> 16) & 0xFFFFu)     // tensor_dim0[31:16]
              | (((unsigned)tile_h & 0xFFFFu) << 16));   // tensor_dim1[15:0]
  g1[3] = (int)((((unsigned)tile_h >> 16) & 0xFFFFu)     // tensor_dim1[31:16]
              | (((unsigned)tile_w & 0xFFFFu) << 16));   // tile_dim0
  g1[4] = (int)((unsigned)tile_h & 0xFFFFu);             // tile_dim1, tile_dim2=0
  g1[5] = (int)(unsigned)row_stride_elems;               // tensor_dim0_stride lo32
  g1[6] = 0;                                             // stride hi | dim1_stride lo
  g1[7] = 0;
  v4i z4; z4[0] = 0; z4[1] = 0; z4[2] = 0; z4[3] = 0;
  v8i z8;
#pragma unroll
  for (int i = 0; i < 8; ++i) z8[i] = 0;
  // 6-arg toolchain form: (g0, g1, g2, g3, g4, cpol)
  __builtin_amdgcn_tensor_load_to_lds(g0, g1, z4, z4, z8, 0);
}

// Load a 16x32 bf16 fragment (A-matrix layout per CDNA5 ISA 7.12.2; identical
// pattern serves as B-frag when the row-major tile holds the transposed
// operand). Two ds_load_b128 per lane.
__device__ __forceinline__ v16bf ld_frag(const __bf16* base, int ldr, int koff) {
  const int l    = threadIdx.x & 31;
  const int row  = l & 15;
  const int hlf  = l >> 4;
  const __bf16* p = base + row * ldr + koff + hlf * 8;
  v8bf c0 = *(const v8bf*)(p);        // K = hlf*8 .. +7
  v8bf c1 = *(const v8bf*)(p + 16);   // K = 16 + hlf*8 .. +7
  v16bf f;
#pragma unroll
  for (int i = 0; i < 8; ++i) { f[i] = c0[i]; f[i + 8] = c1[i]; }
  return f;
}

__device__ __forceinline__ v8f wmma_bf16(v16bf a, v16bf b, v8f c) {
  return __builtin_amdgcn_wmma_f32_16x16x32_bf16(
      /*neg_a=*/false, a, /*neg_b=*/false, b,
      /*c_mod=*/(short)0, c, /*reuse_a=*/false, /*reuse_b=*/false);
}

// ---------------------------------------------------------------------------
// Generic  Out[M,N] = A[M,K] @ W[N,K]^T + bias[N]
// Block: 128x128 tile, BK=32, 256 threads (8 waves, 4x2 wave grid).
// bf16 A tiles are staged via TDM; f32 operands convert through VGPRs.
// ---------------------------------------------------------------------------
template <bool IN_BF16, bool OUT_BF16>
__global__ __launch_bounds__(256, 2)
void gemm_xwT_bias(const void* __restrict__ Ain,
                   const float* __restrict__ W,
                   const float* __restrict__ bias,
                   void* __restrict__ Out,
                   int M, int N, int K)
{
  __shared__ __align__(16) __bf16 sA[128 * 40];  // padded rows: 80B stride
  __shared__ __align__(16) __bf16 sW[128 * 40];

  const int t    = threadIdx.x;
  const int wid  = t >> 5;
  const int lane = t & 31;
  const int hlf  = lane >> 4;
  const int nl   = lane & 15;
  const int wm   = wid & 3;        // wave row   (0..3) -> 32 rows each
  const int wn   = wid >> 2;       // wave col   (0..1) -> 64 cols each
  const int bm   = blockIdx.x;
  const int bn   = blockIdx.y;

  const int lr = t >> 1;           // tile row handled by this thread (0..127)
  const int lc = (t & 1) * 16;     // 16-element half-row

  const float*  Af = (const float*)Ain;
  const __bf16* Ab = (const __bf16*)Ain;

  v8f acc[2][4];
#pragma unroll
  for (int i = 0; i < 2; ++i)
#pragma unroll
    for (int j = 0; j < 4; ++j)
#pragma unroll
      for (int v = 0; v < 8; ++v) acc[i][j][v] = 0.0f;

  for (int kk = 0; kk < K; kk += 32) {
    // -- stage A tile --
    if constexpr (IN_BF16) {
      if (wid == 0)   // TDM ignores EXEC; one wave issues the tile DMA
        tdm_load_2d_bf16((unsigned)(uintptr_t)&sA[0],
                         Ab + (size_t)(bm * 128) * K + kk,
                         /*tile_w=*/32, /*tile_h=*/128, /*stride=*/K,
                         /*pad: every 16 dw*/3, /*4 dw*/3);   // LDS stride 40
      if (kk + 32 < K) __builtin_prefetch(Ab + (size_t)(bm * 128 + lr) * K + kk + 32 + lc, 0, 1);
    } else {
      size_t g = (size_t)(bm * 128 + lr) * K + kk + lc;
      float4 fv[4];
#pragma unroll
      for (int q = 0; q < 4; ++q) fv[q] = *(const float4*)(Af + g + q * 4);
      const float* tf = (const float*)fv;
      v8bf a0, a1;
#pragma unroll
      for (int e = 0; e < 8; ++e) { a0[e] = (__bf16)tf[e]; a1[e] = (__bf16)tf[e + 8]; }
      *(v8bf*)(&sA[lr * 40 + lc])     = a0;
      *(v8bf*)(&sA[lr * 40 + lc + 8]) = a1;
      if (kk + 32 < K) __builtin_prefetch(Af + g + 32, 0, 1);
    }
    // -- stage W tile (rows of W are B-matrix columns of W^T), f32 -> bf16 --
    {
      size_t g = (size_t)(bn * 128 + lr) * K + kk + lc;
      float4 fv[4];
#pragma unroll
      for (int q = 0; q < 4; ++q) fv[q] = *(const float4*)(W + g + q * 4);
      const float* tf = (const float*)fv;
      v8bf a0, a1;
#pragma unroll
      for (int e = 0; e < 8; ++e) { a0[e] = (__bf16)tf[e]; a1[e] = (__bf16)tf[e + 8]; }
      *(v8bf*)(&sW[lr * 40 + lc])     = a0;
      *(v8bf*)(&sW[lr * 40 + lc + 8]) = a1;
      if (kk + 32 < K) __builtin_prefetch(W + g + 32, 0, 1);
    }
    if constexpr (IN_BF16) {
      if (wid == 0) __builtin_amdgcn_s_wait_tensorcnt(0);
    }
    __syncthreads();

    v16bf af[2], bf[4];
#pragma unroll
    for (int i = 0; i < 2; ++i) af[i] = ld_frag(sA + (wm * 32 + i * 16) * 40, 40, 0);
#pragma unroll
    for (int j = 0; j < 4; ++j) bf[j] = ld_frag(sW + (wn * 64 + j * 16) * 40, 40, 0);
#pragma unroll
    for (int i = 0; i < 2; ++i)
#pragma unroll
      for (int j = 0; j < 4; ++j)
        acc[i][j] = wmma_bf16(af[i], bf[j], acc[i][j]);
    __syncthreads();
  }

  // epilogue: bias add, store (C/D layout: reg v -> row v + 8*hlf, lane&15 -> col)
#pragma unroll
  for (int j = 0; j < 4; ++j) {
    const int gn = bn * 128 + wn * 64 + j * 16 + nl;
    const float bj = bias[gn];
#pragma unroll
    for (int i = 0; i < 2; ++i) {
#pragma unroll
      for (int v = 0; v < 8; ++v) {
        const int gm = bm * 128 + wm * 32 + i * 16 + v + hlf * 8;
        const float val = acc[i][j][v] + bj;
        if (OUT_BF16) ((__bf16*)Out)[(size_t)gm * N + gn] = (__bf16)val;
        else          ((float*)Out)[(size_t)gm * N + gn]  = val;
      }
    }
  }
}

// ---------------------------------------------------------------------------
// Flash attention with q=k=v=P. One block = 128 q-rows of one (b,h).
// 8 waves; each wave owns 16 q-rows (full 16x64 output tile in 4 accums).
// Q and KV tiles arrive in LDS via TDM (TENSORcnt), V is transposed LDS->LDS.
// ---------------------------------------------------------------------------
__global__ __launch_bounds__(256, 2)
void attn_kernel(const __bf16* __restrict__ P, __bf16* __restrict__ O)
{
  __shared__ __align__(16) __bf16 sQ[128 * 72];     // 128 x 64, stride 72 (TDM pad)
  __shared__ __align__(16) __bf16 sK[32 * 72];      // [key][dk], stride 72 (TDM pad)
  __shared__ __align__(16) __bf16 sVT[64 * 40];     // [dk][key] (transposed)
  __shared__ __align__(16) __bf16 sP[8][16 * 40];   // per-wave P scratch

  const int t    = threadIdx.x;
  const int wid  = t >> 5;
  const int lane = t & 31;
  const int hlf  = lane >> 4;
  const int nl   = lane & 15;

  const int bh    = blockIdx.y;           // b*H + h
  const int b     = bh >> 4;
  const int h     = bh & 15;
  const int qbase = blockIdx.x * 128;

  // ---- stage Q tile (128 x 64) via TDM: LDS stride 72 ----
  if (wid == 0) {
    tdm_load_2d_bf16((unsigned)(uintptr_t)&sQ[0],
                     P + (size_t)(b * Sc + qbase) * Dc + h * DKc,
                     /*tile_w=*/64, /*tile_h=*/128, /*stride=*/Dc,
                     /*pad: every 32 dw*/4, /*4 dw*/3);
    __builtin_amdgcn_s_wait_tensorcnt(0);
  }
  __syncthreads();

  v16bf aq[2];
  aq[0] = ld_frag(sQ + (wid * 16) * 72, 72, 0);
  aq[1] = ld_frag(sQ + (wid * 16) * 72, 72, 32);

  float m_i[8], l_i[8];
  v8f acc[4];
#pragma unroll
  for (int v = 0; v < 8; ++v) { m_i[v] = -3.0e38f; l_i[v] = 0.0f; }
#pragma unroll
  for (int j = 0; j < 4; ++j)
#pragma unroll
    for (int v = 0; v < 8; ++v) acc[j][v] = 0.0f;

  const float scale = 0.125f;    // 1/sqrt(64)

  for (int kb = 0; kb < Sc; kb += 32) {
    // ---- stage KV chunk (32 keys x 64 dk) via TDM ----
    if (wid == 0) {
      tdm_load_2d_bf16((unsigned)(uintptr_t)&sK[0],
                       P + (size_t)(b * Sc + kb) * Dc + h * DKc,
                       /*tile_w=*/64, /*tile_h=*/32, /*stride=*/Dc,
                       /*pad: every 32 dw*/4, /*4 dw*/3);
      __builtin_amdgcn_s_wait_tensorcnt(0);
    }
    if (kb + 32 < Sc)
      __builtin_prefetch(P + (size_t)(b * Sc + kb + 32 + (t >> 3)) * Dc
                           + h * DKc + (t & 7) * 8, 0, 1);
    __syncthreads();

    // ---- LDS->LDS transpose pass for V columns ----
    {
      const int key = t >> 3;        // 0..31
      const int off = (t & 7) * 8;   // 0..56
      v8bf kv = *(const v8bf*)(&sK[key * 72 + off]);
#pragma unroll
      for (int e = 0; e < 8; ++e) sVT[(off + e) * 40 + key] = kv[e];
    }
    __syncthreads();

    // ---- scores: Q(16x64) @ K^T(64x32) as two 16x16 subtiles, K-loop of 2 --
    v8f s0, s1;
#pragma unroll
    for (int v = 0; v < 8; ++v) { s0[v] = 0.0f; s1[v] = 0.0f; }
    s0 = wmma_bf16(aq[0], ld_frag(sK,           72, 0),  s0);
    s0 = wmma_bf16(aq[1], ld_frag(sK,           72, 32), s0);
    s1 = wmma_bf16(aq[0], ld_frag(sK + 16 * 72, 72, 0),  s1);
    s1 = wmma_bf16(aq[1], ld_frag(sK + 16 * 72, 72, 32), s1);

    // ---- online softmax over the 32-key chunk ----
    __bf16* pw = sP[wid];
#pragma unroll
    for (int v = 0; v < 8; ++v) {
      float a0 = s0[v] * scale;
      float a1 = s1[v] * scale;
      float rm = fmaxf(a0, a1);
#pragma unroll
      for (int msk = 8; msk >= 1; msk >>= 1) rm = fmaxf(rm, __shfl_xor(rm, msk, 16));
      const float mnew = fmaxf(m_i[v], rm);
      const float p0 = __expf(a0 - mnew);
      const float p1 = __expf(a1 - mnew);
      float rs = p0 + p1;
#pragma unroll
      for (int msk = 8; msk >= 1; msk >>= 1) rs += __shfl_xor(rs, msk, 16);
      const float corr = __expf(m_i[v] - mnew);
      l_i[v] = l_i[v] * corr + rs;
      m_i[v] = mnew;
#pragma unroll
      for (int j = 0; j < 4; ++j) acc[j][v] *= corr;
      const int row = v + hlf * 8;
      pw[row * 40 + nl]      = (__bf16)p0;   // keys 0..15
      pw[row * 40 + 16 + nl] = (__bf16)p1;   // keys 16..31
    }

    // ---- O += P(16x32) @ V(32x64); V columns come from transposed tile ----
    v16bf pf = ld_frag(pw, 40, 0);
#pragma unroll
    for (int j = 0; j < 4; ++j)
      acc[j] = wmma_bf16(pf, ld_frag(sVT + (j * 16) * 40, 40, 0), acc[j]);

    __syncthreads();
  }

  // ---- epilogue: normalize, store O in [b,h,s,dk] contiguous order ----
#pragma unroll
  for (int j = 0; j < 4; ++j) {
#pragma unroll
    for (int v = 0; v < 8; ++v) {
      const int m    = v + hlf * 8;
      const int srow = qbase + wid * 16 + m;
      const int dk   = j * 16 + nl;
      const float o  = acc[j][v] / l_i[v];
      const size_t off = (size_t)b * Sc * Dc + (size_t)h * Sc * DKc
                       + (size_t)srow * DKc + dk;
      O[off] = (__bf16)o;
    }
  }
}

// ---------------------------------------------------------------------------
extern "C" void kernel_launch(void* const* d_in, const int* in_sizes, int n_in,
                              void* d_out, int out_size, void* d_ws, size_t ws_size,
                              hipStream_t stream) {
  const float* x  = (const float*)d_in[0];   // [4,2048,1024]
  const float* Wv = (const float*)d_in[1];   // [1024,1024]
  const float* bv = (const float*)d_in[2];   // [1024]
  const float* Wo = (const float*)d_in[3];   // [1024,1024]
  const float* bo = (const float*)d_in[4];   // [1024]
  float* out = (float*)d_out;                // [4,2048,1024] f32

  __bf16* Pbuf = (__bf16*)d_ws;                          // 16 MB
  __bf16* Obuf = Pbuf + (size_t)BSc * Dc;                // 16 MB

  const dim3 blk(256);
  const dim3 gGemm(BSc / 128, Dc / 128);                 // (64, 8)
  const dim3 gAttn(Sc / 128, Bc * Hc);                   // (16, 64)

  // P = x @ Wv^T + bv   (f32 -> bf16)
  gemm_xwT_bias<false, true><<<gGemm, blk, 0, stream>>>(
      (const void*)x, Wv, bv, (void*)Pbuf, BSc, Dc, Dc);

  // O = attention(P)    (bf16 -> bf16, [b,h,s,dk] order == buggy reshape)
  attn_kernel<<<gAttn, blk, 0, stream>>>(Pbuf, Obuf);

  // Y = O @ Wo^T + bo   (bf16 -> f32)
  gemm_xwT_bias<true, false><<<gGemm, blk, 0, stream>>>(
      (const void*)Obuf, Wo, bo, (void*)out, BSc, Dc, Dc);
}
